// TreePool_76699525972511
// MI455X (gfx1250) — matrix operand order repaired and checked by reading, not compiled, not run on hardware
//
#include <hip/hip_runtime.h>

// ---------------------------------------------------------------------------
// TreePool (Tree-LSTM cost model) for MI455X / gfx1250, wave32 + WMMA bf16.
// All GEMMs use v_wmma_f32_16x16x32_bf16 (f32 accumulate). Tree recurrence is
// processed level-parallel (perfect binary tree, 13 levels).
// ---------------------------------------------------------------------------

typedef __attribute__((ext_vector_type(16))) __bf16 v16bf;
typedef __attribute__((ext_vector_type(8)))  __bf16 v8bf;
typedef __attribute__((ext_vector_type(8)))  float  v8f;
typedef __attribute__((ext_vector_type(4)))  float  v4f;

#define N_NODES 8191
#define NROWS_PRED (N_NODES * 8)     // 65528
#define D_IN  576
#define HID   256
#define GATES 1024
#define PDIM  128
#define BMAP  1000
#define BMAPP 1024

__device__ __forceinline__ v8f zero8() {
  v8f v;
#pragma unroll
  for (int i = 0; i < 8; ++i) v[i] = 0.0f;
  return v;
}
__device__ __forceinline__ v16bf join8(v8bf lo, v8bf hi) {
  return __builtin_shufflevector(lo, hi, 0,1,2,3,4,5,6,7,8,9,10,11,12,13,14,15);
}

// A fragment (16x32 bf16): lane half h reads K {k0+8h..+7, k0+8h+16..+7}.
// `base` must be pre-offset by half*8.
__device__ __forceinline__ v16bf loadA_bf16(const __bf16* __restrict__ base, int k0) {
  const v8bf* p0 = reinterpret_cast<const v8bf*>(base + k0);
  const v8bf* p1 = reinterpret_cast<const v8bf*>(base + k0 + 16);
  return join8(*p0, *p1);
}
// B fragment (32x16 bf16): lane half h reads K {k0+8h..+7, k0+8h+16..+7}.
// `base` must be pre-offset by half*8.
__device__ __forceinline__ v16bf loadB_bf16(const __bf16* __restrict__ base, int k0) {
  const v8bf* p0 = reinterpret_cast<const v8bf*>(base + k0);
  const v8bf* p1 = reinterpret_cast<const v8bf*>(base + k0 + 16);
  return join8(*p0, *p1);
}

// A fragment from f32 source, unconditional vector loads (caller guarantees the
// 2x8-float chunks at k0 and k0+16 are in-bounds). `base` pre-offset by half*8.
__device__ __forceinline__ v16bf loadA_f32_fast(const float* __restrict__ base, int k0) {
  v4f f0 = *reinterpret_cast<const v4f*>(base + k0);
  v4f f1 = *reinterpret_cast<const v4f*>(base + k0 + 4);
  v4f f2 = *reinterpret_cast<const v4f*>(base + k0 + 16);
  v4f f3 = *reinterpret_cast<const v4f*>(base + k0 + 20);
  v16bf v;
#pragma unroll
  for (int i = 0; i < 4; ++i) {
    v[i]      = (__bf16)f0[i];
    v[4 + i]  = (__bf16)f1[i];
    v[8 + i]  = (__bf16)f2[i];
    v[12 + i] = (__bf16)f3[i];
  }
  return v;
}
// Branch-free K tail: unconditional loads at clamped index, then select-zero.
__device__ __forceinline__ v16bf loadA_f32_tail(const float* __restrict__ base, int k0, int kmax) {
  v16bf v;
#pragma unroll
  for (int i = 0; i < 8; ++i) {
    int k = k0 + i;
    float f = base[(k < kmax) ? k : 0];
    v[i] = (__bf16)((k < kmax) ? f : 0.0f);
  }
#pragma unroll
  for (int i = 0; i < 8; ++i) {
    int k = k0 + 16 + i;
    float f = base[(k < kmax) ? k : 0];
    v[8 + i] = (__bf16)((k < kmax) ? f : 0.0f);
  }
  return v;
}

#define WMMA_BF16(a, b, c) \
  __builtin_amdgcn_wmma_f32_16x16x32_bf16(false, (a), false, (b), (short)0, (c), false, false)

__device__ __forceinline__ float sigm(float x) { return 1.0f / (1.0f + expf(-x)); }
__device__ __forceinline__ int imin(int a, int b) { return a < b ? a : b; }

// Post-order index of node at height t (0=leaf .. 12=root), in-level position j.
__device__ __forceinline__ int node_index(int t, int j) {
  int idx = (1 << (t + 1)) - 2;   // post-order index of leftmost height-t node
  int base = 0;
  for (int d = t + 1; d <= 12; ++d)
    if ((j >> (d - t - 1)) & 1) base += (1 << d) - 1;
  return base + idx;
}

// ---------------------------------------------------------------------------
// f32 -> bf16 convert with optional column zero-pad (row-major).
__global__ void cvt_pad_kernel(const float* __restrict__ src, __bf16* __restrict__ dst,
                               int rows, int cols, int dcols) {
  int idx = blockIdx.x * blockDim.x + threadIdx.x;
  if (idx >= rows * dcols) return;
  int r = idx / dcols, c = idx - r * dcols;
  dst[idx] = (__bf16)((c < cols) ? src[r * cols + c] : 0.0f);
}

// op_feats -> xb[:, 0:64] (bf16, stride D_IN)
__global__ void opfeat_kernel(const float* __restrict__ op, __bf16* __restrict__ xb) {
  int idx = blockIdx.x * blockDim.x + threadIdx.x;
  if (idx >= N_NODES * 64) return;
  int n = idx >> 6, c = idx & 63;
  xb[n * D_IN + c] = (__bf16)op[idx];
}

// ---------------------------------------------------------------------------
// Predicate embed: out[n*8+p, e] = pred[n,p,:] . Wp[e,:]; min over p fused in
// epilogue (lane half h holds the 8 rows of node 2*tile+h). Writes bf16 to xb.
// Block = 64 threads (2 waves); wave w owns n-group w (64 cols each).
__global__ __launch_bounds__(64) void pred_embed_kernel(
    const float* __restrict__ pred, const __bf16* __restrict__ WpB,
    const float* __restrict__ bp, __bf16* __restrict__ xb, int xoff) {
  const int lane = threadIdx.x & 31, half = lane >> 4, l16 = lane & 15;
  const int m0 = blockIdx.x * 16;                 // row tile in [NROWS_PRED]
  const int n0g = (threadIdx.x >> 5) * 64;        // 4 n-tiles per wave
  // Clamp row (invalid rows compute garbage that is never stored).
  const int arow = imin(m0 + l16, NROWS_PRED - 1);
  const float* aptr = pred + (size_t)arow * PDIM + half * 8;

  const __bf16* bptr[4];
#pragma unroll
  for (int nt = 0; nt < 4; ++nt)
    bptr[nt] = WpB + (size_t)(n0g + nt * 16 + l16) * PDIM + half * 8;

  v8f acc[4];
#pragma unroll
  for (int nt = 0; nt < 4; ++nt) acc[nt] = zero8();

#pragma unroll
  for (int kt = 0; kt < PDIM / 32; ++kt) {
    int k0 = kt * 32;
    v16bf a = loadA_f32_fast(aptr, k0);
#pragma unroll
    for (int nt = 0; nt < 4; ++nt)
      acc[nt] = WMMA_BF16(a, loadB_bf16(bptr[nt], k0), acc[nt]);
  }
  int node = (m0 >> 3) + half;  // rows 0..7 -> node0, rows 8..15 -> node1
  if (node < N_NODES) {
#pragma unroll
    for (int nt = 0; nt < 4; ++nt) {
      float mv = acc[nt][0];
#pragma unroll
      for (int r = 1; r < 8; ++r) mv = fminf(mv, acc[nt][r]);
      int col = n0g + nt * 16 + l16;
      xb[(size_t)node * D_IN + xoff + col] = (__bf16)(mv + bp[col]);
    }
  }
}

// ---------------------------------------------------------------------------
// Bitmap embed: bm = (bitmap @ Wb^T + bb) * has_cond  -> xb[:, 320:576] (bf16)
// Block = 128 threads (4 waves); wave w owns n-group w.
__global__ __launch_bounds__(128) void bitmap_embed_kernel(
    const float* __restrict__ bitmap, const __bf16* __restrict__ WbB,
    const float* __restrict__ bb, const int* __restrict__ has_cond,
    __bf16* __restrict__ xb) {
  const int lane = threadIdx.x & 31, half = lane >> 4, l16 = lane & 15;
  const int m0 = blockIdx.x * 16;
  const int n0g = (threadIdx.x >> 5) * 64;
  const int arow = imin(m0 + l16, N_NODES - 1);
  const float* aptr = bitmap + (size_t)arow * BMAP + half * 8;

  const __bf16* bptr[4];
#pragma unroll
  for (int nt = 0; nt < 4; ++nt)
    bptr[nt] = WbB + (size_t)(n0g + nt * 16 + l16) * BMAPP + half * 8;

  v8f acc[4];
#pragma unroll
  for (int nt = 0; nt < 4; ++nt) acc[nt] = zero8();

  // Fast K-steps: largest element index = k0 + 8 + 16 + 7 <= 991 < 1000.
  for (int kt = 0; kt < BMAPP / 32 - 1; ++kt) {
    int k0 = kt * 32;
    v16bf a = loadA_f32_fast(aptr, k0);
#pragma unroll
    for (int nt = 0; nt < 4; ++nt)
      acc[nt] = WMMA_BF16(a, loadB_bf16(bptr[nt], k0), acc[nt]);
  }
  {  // Tail K-step (k = 992..1023, valid < 1000), branch-free clamped loads.
    int k0 = BMAPP - 32;
    v16bf a = loadA_f32_tail(aptr, k0, BMAP - half * 8);
#pragma unroll
    for (int nt = 0; nt < 4; ++nt)
      acc[nt] = WMMA_BF16(a, loadB_bf16(bptr[nt], k0), acc[nt]);
  }
#pragma unroll
  for (int nt = 0; nt < 4; ++nt) {
    int col = n0g + nt * 16 + l16;
    float bbc = bb[col];
#pragma unroll
    for (int r = 0; r < 8; ++r) {
      int row = m0 + r + half * 8;
      if (row < N_NODES) {
        float v = (acc[nt][r] + bbc) * (float)has_cond[row];
        xb[(size_t)row * D_IN + 320 + col] = (__bf16)v;
      }
    }
  }
}

// ---------------------------------------------------------------------------
// pre = x @ Wih^T + bih + bhh   ([8191,576] x [576,1024] -> f32 [8191,1024])
// Block = 128 threads (4 waves); grid.y covers 4 groups of 4 n-groups.
__global__ __launch_bounds__(128) void x_gemm_kernel(
    const __bf16* __restrict__ xb, const __bf16* __restrict__ WihB,
    const float* __restrict__ bih, const float* __restrict__ bhh,
    float* __restrict__ pre) {
  const int lane = threadIdx.x & 31, half = lane >> 4, l16 = lane & 15;
  const int m0 = blockIdx.x * 16;
  const int n0g = (blockIdx.y * 4 + (threadIdx.x >> 5)) * 64;
  const int arow = imin(m0 + l16, N_NODES - 1);
  const __bf16* aptr = xb + (size_t)arow * D_IN + half * 8;

  const __bf16* bptr[4];
#pragma unroll
  for (int nt = 0; nt < 4; ++nt)
    bptr[nt] = WihB + (size_t)(n0g + nt * 16 + l16) * D_IN + half * 8;

  v8f acc[4];
#pragma unroll
  for (int nt = 0; nt < 4; ++nt) acc[nt] = zero8();

  for (int kt = 0; kt < D_IN / 32; ++kt) {
    int k0 = kt * 32;
    v16bf a = loadA_bf16(aptr, k0);
#pragma unroll
    for (int nt = 0; nt < 4; ++nt)
      acc[nt] = WMMA_BF16(a, loadB_bf16(bptr[nt], k0), acc[nt]);
  }
#pragma unroll
  for (int nt = 0; nt < 4; ++nt) {
    int col = n0g + nt * 16 + l16;
    float bc = bih[col] + bhh[col];
#pragma unroll
    for (int r = 0; r < 8; ++r) {
      int row = m0 + r + half * 8;
      if (row < N_NODES) pre[(size_t)row * GATES + col] = acc[nt][r] + bc;
    }
  }
}

// ---------------------------------------------------------------------------
// Per-level: gather children -> havg (bf16), cavg (f32)
__global__ void gather_kernel(const float* __restrict__ Hs, const float* __restrict__ Cs,
                              const int* __restrict__ left, const int* __restrict__ right,
                              int t, int nodes, __bf16* __restrict__ havg,
                              float* __restrict__ cavg) {
  int idx = blockIdx.x * blockDim.x + threadIdx.x;
  if (idx >= nodes * HID) return;
  int m = idx >> 8, j = idx & 255;
  int node = node_index(t, m);
  int l = left[node], r = right[node];
  havg[idx] = (__bf16)(0.5f * (Hs[l * HID + j] + Hs[r * HID + j]));
  cavg[idx] = 0.5f * (Cs[l * HID + j] + Cs[r * HID + j]);
}

// gbuf = havg @ Whh^T  ([nodes,256] x [256,1024]); nodes multiple of 16.
// Block = 128 threads (4 waves); grid.y covers 4 groups of 4 n-groups.
__global__ __launch_bounds__(128) void level_gemm_kernel(
    const __bf16* __restrict__ havg, const __bf16* __restrict__ WhhB,
    float* __restrict__ gbuf) {
  const int lane = threadIdx.x & 31, half = lane >> 4, l16 = lane & 15;
  const int m0 = blockIdx.x * 16;
  const int n0g = (blockIdx.y * 4 + (threadIdx.x >> 5)) * 64;
  const __bf16* aptr = havg + (size_t)(m0 + l16) * HID + half * 8;

  const __bf16* bptr[4];
#pragma unroll
  for (int nt = 0; nt < 4; ++nt)
    bptr[nt] = WhhB + (size_t)(n0g + nt * 16 + l16) * HID + half * 8;

  v8f acc[4];
#pragma unroll
  for (int nt = 0; nt < 4; ++nt) acc[nt] = zero8();

#pragma unroll
  for (int kt = 0; kt < HID / 32; ++kt) {
    int k0 = kt * 32;
    v16bf a = loadA_bf16(aptr, k0);
#pragma unroll
    for (int nt = 0; nt < 4; ++nt)
      acc[nt] = WMMA_BF16(a, loadB_bf16(bptr[nt], k0), acc[nt]);
  }
#pragma unroll
  for (int nt = 0; nt < 4; ++nt) {
    int col = n0g + nt * 16 + l16;
#pragma unroll
    for (int r = 0; r < 8; ++r)
      gbuf[(size_t)(m0 + r + half * 8) * GATES + col] = acc[nt][r];
  }
}

// LSTM cell: g = pre[node] (+ gbuf[m]); torch gate order i,f,g,o.
__global__ void gate_kernel(const float* __restrict__ pre, const float* __restrict__ gbuf,
                            const float* __restrict__ cavg, int t, int nodes,
                            float* __restrict__ Hs, float* __restrict__ Cs) {
  int idx = blockIdx.x * blockDim.x + threadIdx.x;
  if (idx >= nodes * HID) return;
  int m = idx >> 8, j = idx & 255;
  int node = node_index(t, m);
  size_t pn = (size_t)node * GATES, gm = (size_t)m * GATES;
  float gi = pre[pn + j]       + (gbuf ? gbuf[gm + j]       : 0.0f);
  float gf = pre[pn + 256 + j] + (gbuf ? gbuf[gm + 256 + j] : 0.0f);
  float gg = pre[pn + 512 + j] + (gbuf ? gbuf[gm + 512 + j] : 0.0f);
  float go = pre[pn + 768 + j] + (gbuf ? gbuf[gm + 768 + j] : 0.0f);
  float c  = cavg ? cavg[idx] : 0.0f;
  float c2 = sigm(gf) * c + sigm(gi) * tanhf(gg);
  Hs[node * HID + j] = sigm(go) * tanhf(c2);
  Cs[node * HID + j] = c2;
}

// Tiny levels (<16 nodes): one block per node, shared-mem matvec in f32.
__global__ __launch_bounds__(256) void lstm_small_kernel(
    const float* __restrict__ pre, const float* __restrict__ Whh,
    const int* __restrict__ left, const int* __restrict__ right, int t,
    float* __restrict__ Hs, float* __restrict__ Cs) {
  __shared__ float sh[HID];
  int node = node_index(t, blockIdx.x);
  int j = threadIdx.x;
  int l = left[node], r = right[node];
  sh[j] = 0.5f * (Hs[l * HID + j] + Hs[r * HID + j]);
  float c = 0.5f * (Cs[l * HID + j] + Cs[r * HID + j]);
  __syncthreads();
  float g[4];
#pragma unroll
  for (int s = 0; s < 4; ++s) {
    int col = s * HID + j;
    float acc = pre[(size_t)node * GATES + col];
    const float* w = Whh + (size_t)col * HID;
#pragma unroll 8
    for (int k = 0; k < HID; ++k) acc += sh[k] * w[k];
    g[s] = acc;
  }
  float c2 = sigm(g[1]) * c + sigm(g[0]) * tanhf(g[2]);
  Hs[node * HID + j] = sigm(g[3]) * tanhf(c2);
  Cs[node * HID + j] = c2;
}

// Output heads: two 3-layer MLPs on root hidden state; writes d_out[0..1].
__global__ __launch_bounds__(256) void head_kernel(
    const float* __restrict__ Hs,
    const float* __restrict__ W2a, const float* __restrict__ b2a,
    const float* __restrict__ W3a, const float* __restrict__ b3a,
    const float* __restrict__ Woa, const float* __restrict__ boa,
    const float* __restrict__ W2b, const float* __restrict__ b2b,
    const float* __restrict__ W3b, const float* __restrict__ b3b,
    const float* __restrict__ Wob, const float* __restrict__ bob,
    float* __restrict__ out) {
  __shared__ float s0[HID], s1[HID], s2[HID];
  int j = threadIdx.x;
  s0[j] = Hs[(N_NODES - 1) * HID + j];
  __syncthreads();

  const float* W2[2] = {W2a, W2b}; const float* b2[2] = {b2a, b2b};
  const float* W3[2] = {W3a, W3b}; const float* b3[2] = {b3a, b3b};
  const float* Wo[2] = {Woa, Wob}; const float* bo[2] = {boa, bob};

  for (int h = 0; h < 2; ++h) {
    float acc = b2[h][j];
    for (int k = 0; k < HID; ++k) acc += s0[k] * W2[h][j * HID + k];
    s1[j] = fmaxf(acc, 0.0f);
    __syncthreads();
    acc = b3[h][j];
    for (int k = 0; k < HID; ++k) acc += s1[k] * W3[h][j * HID + k];
    s2[j] = fmaxf(acc, 0.0f);
    __syncthreads();
    s1[j] = s2[j] * Wo[h][j];
    __syncthreads();
    for (int off = 128; off > 0; off >>= 1) {
      if (j < off) s1[j] += s1[j + off];
      __syncthreads();
    }
    if (j == 0) out[h] = sigm(s1[0] + bo[h][0]);
    __syncthreads();
  }
}

// ---------------------------------------------------------------------------
extern "C" void kernel_launch(void* const* d_in, const int* in_sizes, int n_in,
                              void* d_out, int out_size, void* d_ws, size_t ws_size,
                              hipStream_t stream) {
  const float* op_feats = (const float*)d_in[0];
  const float* pred1    = (const float*)d_in[1];
  const float* pred2    = (const float*)d_in[2];
  const float* bitmap   = (const float*)d_in[3];
  const int*   has_cond = (const int*)d_in[4];
  const int*   left     = (const int*)d_in[5];
  const int*   right    = (const int*)d_in[6];
  const float* Wp  = (const float*)d_in[7];
  const float* bp  = (const float*)d_in[8];
  const float* Wb  = (const float*)d_in[9];
  const float* bb  = (const float*)d_in[10];
  const float* Wih = (const float*)d_in[11];
  const float* Whh = (const float*)d_in[12];
  const float* bih = (const float*)d_in[13];
  const float* bhh = (const float*)d_in[14];
  const float* W2a = (const float*)d_in[15];
  const float* b2a = (const float*)d_in[16];
  const float* W3a = (const float*)d_in[17];
  const float* b3a = (const float*)d_in[18];
  const float* Woa = (const float*)d_in[19];
  const float* boa = (const float*)d_in[20];
  const float* W2b = (const float*)d_in[21];
  const float* b2b = (const float*)d_in[22];
  const float* W3b = (const float*)d_in[23];
  const float* b3b = (const float*)d_in[24];
  const float* Wob = (const float*)d_in[25];
  const float* bob = (const float*)d_in[26];
  float* out = (float*)d_out;

  // Workspace carve-up (~74 MB total)
  char* w = (char*)d_ws;
  auto carve = [&](size_t bytes) -> void* {
    void* p = (void*)w;
    w += (bytes + 255) & ~(size_t)255;
    return p;
  };
  __bf16* WpB  = (__bf16*)carve((size_t)PDIM * PDIM * 2);
  __bf16* WbB  = (__bf16*)carve((size_t)HID * BMAPP * 2);
  __bf16* WihB = (__bf16*)carve((size_t)GATES * D_IN * 2);
  __bf16* WhhB = (__bf16*)carve((size_t)GATES * HID * 2);
  __bf16* xb   = (__bf16*)carve((size_t)N_NODES * D_IN * 2);
  float*  pre  = (float*)carve((size_t)N_NODES * GATES * 4);
  float*  Hs   = (float*)carve((size_t)N_NODES * HID * 4);
  float*  Cs   = (float*)carve((size_t)N_NODES * HID * 4);
  __bf16* havg = (__bf16*)carve((size_t)2048 * HID * 2);
  float*  cavg = (float*)carve((size_t)2048 * HID * 4);
  float*  gbuf = (float*)carve((size_t)2048 * GATES * 4);

  auto blks = [](int n) { return (n + 255) / 256; };

  // Weight conversion (bf16, Wb zero-padded K 1000 -> 1024)
  cvt_pad_kernel<<<blks(PDIM * PDIM), 256, 0, stream>>>(Wp, WpB, PDIM, PDIM, PDIM);
  cvt_pad_kernel<<<blks(HID * BMAPP), 256, 0, stream>>>(Wb, WbB, HID, BMAP, BMAPP);
  cvt_pad_kernel<<<blks(GATES * D_IN), 256, 0, stream>>>(Wih, WihB, GATES, D_IN, D_IN);
  cvt_pad_kernel<<<blks(GATES * HID), 256, 0, stream>>>(Whh, WhhB, GATES, HID, HID);

  // Assemble x (bf16): op | cond1 | cond2 | bitmap-embed
  opfeat_kernel<<<blks(N_NODES * 64), 256, 0, stream>>>(op_feats, xb);
  pred_embed_kernel<<<4096, 64, 0, stream>>>(pred1, WpB, bp, xb, 64);
  pred_embed_kernel<<<4096, 64, 0, stream>>>(pred2, WpB, bp, xb, 192);
  bitmap_embed_kernel<<<512, 128, 0, stream>>>(bitmap, WbB, bb, has_cond, xb);

  // pre = x @ Wih^T + bih + bhh for all nodes (recurrence-independent)
  x_gemm_kernel<<<dim3(512, 4), 128, 0, stream>>>(xb, WihB, bih, bhh, pre);

  // Level 0 (leaves): h = c = 0 -> gates directly from pre
  gate_kernel<<<blks(4096 * HID), 256, 0, stream>>>(pre, nullptr, nullptr, 0, 4096, Hs, Cs);

  // Levels 1..8 (2048 .. 16 nodes): WMMA path
  for (int t = 1; t <= 8; ++t) {
    int nodes = 4096 >> t;
    gather_kernel<<<blks(nodes * HID), 256, 0, stream>>>(Hs, Cs, left, right, t, nodes, havg, cavg);
    level_gemm_kernel<<<dim3(nodes / 16, 4), 128, 0, stream>>>(havg, WhhB, gbuf);
    gate_kernel<<<blks(nodes * HID), 256, 0, stream>>>(pre, gbuf, cavg, t, nodes, Hs, Cs);
  }

  // Levels 9..12 (8,4,2,1 nodes): shared-memory matvec
  for (int t = 9; t <= 12; ++t) {
    int nodes = 4096 >> t;
    lstm_small_kernel<<<nodes, 256, 0, stream>>>(pre, Whh, left, right, t, Hs, Cs);
  }

  // Output heads on root hidden state
  head_kernel<<<1, 256, 0, stream>>>(Hs, W2a, b2a, W3a, b3a, Woa, boa,
                                     W2b, b2b, W3b, b3b, Wob, bob, out);
}
